// GAT_62654982914445
// MI455X (gfx1250) — compile-verified, matrix operand order checked by reading, not run on hardware
//
#include <hip/hip_runtime.h>
#include <hip/hip_bf16.h>
#include <math.h>

// ---------------------------------------------------------------------------
// GAT forward for MI455X (gfx1250, wave32).
//   B=2, N=2048, E=32768, Fin=256, nhead=8, nhid=8 (concat 64), Fout=64.
// Pipeline:
//   1) cvt x -> f16, pack W_h heads -> W1t[64][256] (transposed), W_o -> W2t[64][64]
//   2) h1 = x @ Wcat      (WMMA f32_16x16x32_f16, f32 accum)
//   3) per-node scores s1,s2 ; per-edge leaky_relu + atomic-max + exp-sum
//   4) edge-weighted aggregation (atomicAdd), uniform fallback for empty rows
//   5) elu -> f16, second GAT layer (same machinery, 1 head of 64)
//   6) log_softmax over the node axis per (batch, feature) column
// ---------------------------------------------------------------------------

typedef __attribute__((ext_vector_type(16))) _Float16 v16h;
typedef __attribute__((ext_vector_type(8)))  _Float16 v8h;
typedef __attribute__((ext_vector_type(8)))  float    v8f;

static constexpr int Bn    = 2;
static constexpr int Nn    = 2048;
static constexpr int En    = 32768;
static constexpr int FIN   = 256;
static constexpr int HF    = 64;          // concat hidden = 8 heads * 8 = layer2 width
static constexpr int NODES = Bn * Nn;     // 4096

// ---- monotone float <-> uint key for atomicMax on floats -------------------
__device__ __forceinline__ unsigned fkey(float v) {
  unsigned u = __float_as_uint(v);
  return (u & 0x80000000u) ? ~u : (u | 0x80000000u);
}
__device__ __forceinline__ float fdec(unsigned k) {
  unsigned u = (k & 0x80000000u) ? (k & 0x7fffffffu) : ~k;
  return __uint_as_float(u);
}

// ---- load a 16-bit WMMA A/B fragment: two contiguous b128 chunks -----------
// lane group g (lane>>4): halves 0..7 = K0+g*8 .. +7 ; halves 8..15 = K0+16+g*8 .. +7
__device__ __forceinline__ v16h load_frag16(const _Float16* __restrict__ p) {
  union { v16h v; struct { v8h lo; v8h hi; } s; } u;
  u.s.lo = *(const v8h*)(p);
  u.s.hi = *(const v8h*)(p + 16);
  return u.v;
}

// ---- GEMM: C[M][64] = A[M][K] (f16, row-major) x Bt[64][K] (f16, B^T) ------
// one wave per (16-row M tile, 16-col N tile); 4 waves per block cover N=64.
template<int K>
__global__ __launch_bounds__(128)
void k_wmma_gemm(const _Float16* __restrict__ A,
                 const _Float16* __restrict__ Bt,
                 float* __restrict__ C) {
  const int lane = threadIdx.x & 31;
  const int nt   = threadIdx.x >> 5;     // N tile 0..3
  const int mt   = blockIdx.x;           // M tile
  const int g    = lane >> 4;
  const int l    = lane & 15;

  const _Float16* ap = A  + (size_t)(mt * 16 + l) * K + g * 8;
  const _Float16* bp = Bt + (size_t)(nt * 16 + l) * K + g * 8;

  v8f acc = {};
#pragma unroll
  for (int k0 = 0; k0 < K; k0 += 32) {
    v16h af = load_frag16(ap + k0);
    v16h bf = load_frag16(bp + k0);
    acc = __builtin_amdgcn_wmma_f32_16x16x32_f16(
        /*neg_a=*/false, af, /*neg_b=*/false, bf,
        /*c_mod=*/(short)0, acc, /*reuse_a=*/false, /*reuse_b=*/false);
  }
  // C layout: VGPR v -> row (mt*16 + g*8 + v), col nt*16 + l
  float* cp = C + (size_t)(mt * 16 + g * 8) * HF + nt * 16 + l;
#pragma unroll
  for (int v = 0; v < 8; v++) cp[(size_t)v * HF] = acc[v];
}

// ---- small utility kernels -------------------------------------------------
__global__ void k_zero_u32(unsigned* __restrict__ p, int n) {
  int i = blockIdx.x * 256 + threadIdx.x;
  if (i < n) p[i] = 0u;
}
__global__ void k_cvt_f16(const float* __restrict__ src, _Float16* __restrict__ dst, int n) {
  int i = blockIdx.x * 256 + threadIdx.x;
  if (i < n) dst[i] = (_Float16)src[i];
}
// W1t[c][k] = W_h[h][k][o], c = h*8+o  (heads concatenated, then transposed)
__global__ void k_prep_w1(const float* __restrict__ W_h, _Float16* __restrict__ W1t) {
  int i = blockIdx.x * 256 + threadIdx.x;
  if (i >= 64 * FIN) return;
  int c = i / FIN, k = i % FIN;
  int h = c >> 3, o = c & 7;
  W1t[i] = (_Float16)W_h[h * (FIN * 8) + k * 8 + o];
}
// W2t[c][k] = W_o[k][c]
__global__ void k_prep_w2(const float* __restrict__ W_o, _Float16* __restrict__ W2t) {
  int i = blockIdx.x * 256 + threadIdx.x;
  if (i >= 64 * 64) return;
  int c = i >> 6, k = i & 63;
  W2t[i] = (_Float16)W_o[k * 64 + c];
}

// s1[n][h] = sum_o h[n][h*8+o]*a[h*16+o];  s2 uses a[h*16+8+o]
__global__ void k_score1(const float* __restrict__ h, const float* __restrict__ a,
                         float* __restrict__ s1, float* __restrict__ s2) {
  int i = blockIdx.x * 256 + threadIdx.x;
  if (i >= NODES * 8) return;
  int n = i >> 3, hh = i & 7;
  const float* hp = h + (size_t)n * HF + hh * 8;
  const float* av = a + hh * 16;
  float s = 0.f, t = 0.f;
#pragma unroll
  for (int o = 0; o < 8; o++) { s += hp[o] * av[o]; t += hp[o] * av[8 + o]; }
  s1[i] = s; s2[i] = t;
}
__global__ void k_score2(const float* __restrict__ h, const float* __restrict__ a,
                         float* __restrict__ s1, float* __restrict__ s2) {
  int n = blockIdx.x * 256 + threadIdx.x;
  if (n >= NODES) return;
  const float* hp = h + (size_t)n * HF;
  float s = 0.f, t = 0.f;
#pragma unroll
  for (int c = 0; c < 64; c++) { s += hp[c] * a[c]; t += hp[c] * a[64 + c]; }
  s1[n] = s; s2[n] = t;
}

// ---- edge passes (H = heads; per-head feature width O = 64/H) --------------
template<int H>
__global__ void k_edge_score_max(const int* __restrict__ edges,
                                 const float* __restrict__ s1, const float* __restrict__ s2,
                                 unsigned* __restrict__ menc, float* __restrict__ ew) {
  int idx = blockIdx.x * 256 + threadIdx.x;
  if (idx >= Bn * En * H) return;
  int hh = idx % H, e = (idx / H) % En, b = idx / (H * En);
  int rs = b * Nn + edges[e];
  int rt = b * Nn + edges[En + e];
  float v = s1[rs * H + hh] + s2[rt * H + hh];
  v = v > 0.f ? v : 0.2f * v;                       // leaky_relu(0.2)
  ew[idx] = v;
  atomicMax(menc + rs * H + hh, fkey(v));
}

template<int H>
__global__ void k_edge_expsum(const int* __restrict__ edges,
                              float* __restrict__ ew,
                              const unsigned* __restrict__ menc,
                              float* __restrict__ Z) {
  int idx = blockIdx.x * 256 + threadIdx.x;
  if (idx >= Bn * En * H) return;
  int hh = idx % H, e = (idx / H) % En, b = idx / (H * En);
  int rs = b * Nn + edges[e];
  float w = expf(ew[idx] - fdec(menc[rs * H + hh]));
  ew[idx] = w;
  atomicAdd(Z + rs * H + hh, w);
}

template<int H>
__global__ void k_edge_agg(const int* __restrict__ edges,
                           const float* __restrict__ ew,
                           const float* __restrict__ h,
                           float* __restrict__ agg) {
  constexpr int O = 64 / H;
  int idx = blockIdx.x * 256 + threadIdx.x;
  if (idx >= Bn * En * H) return;
  int hh = idx % H, e = (idx / H) % En, b = idx / (H * En);
  int rs = b * Nn + edges[e];
  int rt = b * Nn + edges[En + e];
  float w = ew[idx];
  const float* hp = h + (size_t)rt * HF + hh * O;
  float* ap = agg + (size_t)rs * HF + hh * O;
#pragma unroll
  for (int o = 0; o < O; o++) atomicAdd(ap + o, w * hp[o]);
}

// column means of h over nodes, per batch (fallback for zero-out-degree rows)
__global__ void k_colmean(const float* __restrict__ h, float* __restrict__ cm) {
  int i = threadIdx.x;                  // 128 = Bn*64
  int b = i >> 6, c = i & 63;
  float s = 0.f;
  for (int n = 0; n < Nn; n++) s += h[(size_t)(b * Nn + n) * HF + c];
  cm[i] = s * (1.f / Nn);
}

// layer-1 finish: normalize, +bias, elu, store f16 for next GEMM
__global__ void k_finish1(const float* __restrict__ agg, const float* __restrict__ Z,
                          const float* __restrict__ cm, const float* __restrict__ bias,
                          _Float16* __restrict__ x1h) {
  int i = blockIdx.x * 256 + threadIdx.x;
  if (i >= NODES * HF) return;
  int row = i >> 6, c = i & 63, hh = c >> 3, b = row >> 11;
  float z = Z[row * 8 + hh];
  float v = (z > 0.f) ? (agg[i] / z) : cm[b * 64 + c];
  v += bias[c];
  v = v > 0.f ? v : expm1f(v);                      // elu
  x1h[i] = (_Float16)v;
}

// layer-2 finish: normalize, +bias, store f32
__global__ void k_finish2(const float* __restrict__ agg, const float* __restrict__ Z,
                          const float* __restrict__ cm, const float* __restrict__ bias,
                          float* __restrict__ y) {
  int i = blockIdx.x * 256 + threadIdx.x;
  if (i >= NODES * HF) return;
  int row = i >> 6, c = i & 63, b = row >> 11;
  float z = Z[row];
  float v = (z > 0.f) ? (agg[i] / z) : cm[b * 64 + c];
  y[i] = v + bias[c];
}

// log_softmax over the node axis: one block per (b, feature) column
__global__ __launch_bounds__(256)
void k_logsoftmax(const float* __restrict__ y, float* __restrict__ out) {
  __shared__ float red[256];
  int b = blockIdx.x >> 6, o = blockIdx.x & 63;
  const float* yp = y + (size_t)b * Nn * HF + o;
  int tid = threadIdx.x;

  float m = -INFINITY;
  for (int n = tid; n < Nn; n += 256) m = fmaxf(m, yp[(size_t)n * HF]);
  red[tid] = m; __syncthreads();
  for (int s = 128; s > 0; s >>= 1) {
    if (tid < s) red[tid] = fmaxf(red[tid], red[tid + s]);
    __syncthreads();
  }
  m = red[0]; __syncthreads();

  float ss = 0.f;
  for (int n = tid; n < Nn; n += 256) ss += expf(yp[(size_t)n * HF] - m);
  red[tid] = ss; __syncthreads();
  for (int s = 128; s > 0; s >>= 1) {
    if (tid < s) red[tid] += red[tid + s];
    __syncthreads();
  }
  float lse = m + logf(red[0]);

  float* op = out + (size_t)b * Nn * HF + o;
  for (int n = tid; n < Nn; n += 256) op[(size_t)n * HF] = yp[(size_t)n * HF] - lse;
}

// ---------------------------------------------------------------------------
extern "C" void kernel_launch(void* const* d_in, const int* in_sizes, int n_in,
                              void* d_out, int out_size, void* d_ws, size_t ws_size,
                              hipStream_t stream) {
  (void)in_sizes; (void)n_in; (void)out_size; (void)ws_size;
  const float* x    = (const float*)d_in[0];
  const int*   edg  = (const int*)  d_in[1];
  const float* W_h  = (const float*)d_in[2];
  const float* a_h  = (const float*)d_in[3];
  const float* b_h  = (const float*)d_in[4];
  const float* W_o  = (const float*)d_in[5];
  const float* a_o  = (const float*)d_in[6];
  const float* b_o  = (const float*)d_in[7];
  float* out = (float*)d_out;

  // bump-allocate workspace (256B aligned); ~7 MB total
  char* w = (char*)d_ws;
  auto alloc = [&](size_t bytes) -> char* {
    char* p = w; w += (bytes + 255) & ~(size_t)255; return p;
  };
  _Float16* xh  = (_Float16*)alloc((size_t)NODES * FIN * 2);  // also reused as y2 (f32, 1MB <= 2MB)
  _Float16* W1t = (_Float16*)alloc((size_t)64 * FIN * 2);
  _Float16* W2t = (_Float16*)alloc((size_t)64 * 64 * 2);
  float*    h   = (float*)   alloc((size_t)NODES * HF * 4);   // h1, then h2
  float*    s1  = (float*)   alloc((size_t)NODES * 8 * 4);
  float*    s2  = (float*)   alloc((size_t)NODES * 8 * 4);
  unsigned* me  = (unsigned*)alloc((size_t)NODES * 8 * 4);
  float*    Z   = (float*)   alloc((size_t)NODES * 8 * 4);
  float*    ew  = (float*)   alloc((size_t)Bn * En * 8 * 4);
  float*    agg = (float*)   alloc((size_t)NODES * HF * 4);
  float*    cm  = (float*)   alloc((size_t)Bn * 64 * 4);
  _Float16* x1h = (_Float16*)alloc((size_t)NODES * HF * 2);
  float*    y2  = (float*)xh;                                  // reuse (xh dead after GEMM1)

  auto g = [](int n) { return (n + 255) / 256; };

  // prep
  k_cvt_f16<<<g(NODES * FIN), 256, 0, stream>>>(x, xh, NODES * FIN);
  k_prep_w1<<<g(64 * FIN), 256, 0, stream>>>(W_h, W1t);
  k_prep_w2<<<g(64 * 64), 256, 0, stream>>>(W_o, W2t);

  // ---- layer 1 (8 heads of 8) ----
  k_wmma_gemm<FIN><<<NODES / 16, 128, 0, stream>>>(xh, W1t, h);
  k_score1<<<g(NODES * 8), 256, 0, stream>>>(h, a_h, s1, s2);
  k_zero_u32<<<g(NODES * 8), 256, 0, stream>>>(me, NODES * 8);
  k_zero_u32<<<g(NODES * 8), 256, 0, stream>>>((unsigned*)Z, NODES * 8);
  k_zero_u32<<<g(NODES * HF), 256, 0, stream>>>((unsigned*)agg, NODES * HF);
  k_edge_score_max<8><<<g(Bn * En * 8), 256, 0, stream>>>(edg, s1, s2, me, ew);
  k_edge_expsum<8><<<g(Bn * En * 8), 256, 0, stream>>>(edg, ew, me, Z);
  k_edge_agg<8><<<g(Bn * En * 8), 256, 0, stream>>>(edg, ew, h, agg);
  k_colmean<<<1, 128, 0, stream>>>(h, cm);
  k_finish1<<<g(NODES * HF), 256, 0, stream>>>(agg, Z, cm, b_h, x1h);

  // ---- layer 2 (1 head of 64) ----
  k_wmma_gemm<HF><<<NODES / 16, 128, 0, stream>>>(x1h, W2t, h);
  k_score2<<<g(NODES), 256, 0, stream>>>(h, a_o, s1, s2);
  k_zero_u32<<<g(NODES), 256, 0, stream>>>(me, NODES);
  k_zero_u32<<<g(NODES), 256, 0, stream>>>((unsigned*)Z, NODES);
  k_zero_u32<<<g(NODES * HF), 256, 0, stream>>>((unsigned*)agg, NODES * HF);
  k_edge_score_max<1><<<g(Bn * En), 256, 0, stream>>>(edg, s1, s2, me, ew);
  k_edge_expsum<1><<<g(Bn * En), 256, 0, stream>>>(edg, ew, me, Z);
  k_edge_agg<1><<<g(Bn * En), 256, 0, stream>>>(edg, ew, h, agg);
  k_colmean<<<1, 128, 0, stream>>>(h, cm);
  k_finish2<<<g(NODES * HF), 256, 0, stream>>>(agg, Z, cm, b_o, y2);

  // ---- log_softmax over nodes ----
  k_logsoftmax<<<Bn * 64, 256, 0, stream>>>(y2, out);
}
